// AttentionDecoder_9294309229274
// MI455X (gfx1250) — compile-verified, matrix operand order checked by reading, not run on hardware
//
#include <hip/hip_runtime.h>

// ---------------------------------------------------------------------------
// AttentionDecoder for MI455X (gfx1250, wave32, WMMA).
//
// Shapes: B=64, L=1024, C=512, H=512, E=80, V=1000, T=200.
//   * One-time: bf16 weight conversion (pad W_ih K 592->608, W_out rows
//     1000->1008), embedding gather, mean/h0/c0/o0, enc_proj WMMA GEMM.
//   * Per step: K-split fused gates-WMMA + LSTM pointwise (LDS reduce);
//     K-split hW2 WMMA; attention score stream (134 MB); softmax; context
//     stream (134 MB, deterministic 2-phase reduce); K-split o-WMMA; fused
//     logits-WMMA + vocab softmax.
// Small GEMMs split their K tiles across waves (LDS partial reduce) to cut
// the serial dependent-load chain (latency-bound at B=64).
// Workspace requirement: ~148 MB.
// ---------------------------------------------------------------------------

typedef __bf16 bf16;
typedef __attribute__((ext_vector_type(16))) __bf16 v16bf;
typedef __attribute__((ext_vector_type(8)))  float  v8f;

#define NB 64
#define NL 1024
#define NC 512
#define NH 512
#define NE 80
#define NV 1000
#define NT 200

__device__ __forceinline__ bf16  f2bf(float x){ return (bf16)x; }
__device__ __forceinline__ float fast_exp(float x){ return __builtin_amdgcn_exp2f(x * 1.4426950408889634f); }
__device__ __forceinline__ float fast_rcp(float x){ return __builtin_amdgcn_rcpf(x); }
__device__ __forceinline__ float fast_sigmoid(float x){ return fast_rcp(1.0f + fast_exp(-x)); }
__device__ __forceinline__ float fast_tanh(float x){
  float ax = fabsf(x);
  float t  = fast_exp(-2.0f * ax);
  float r  = (1.0f - t) * fast_rcp(1.0f + t);
  return copysignf(r, x);
}

// K offset inside a 32-wide K tile for 16-bit A/B fragments (ISA 7.12.2):
// VGPR v: k = (v>>2)*16 + half*8 + (v&3)*2  (pairs are contiguous).
__device__ __forceinline__ int kfoff(int v, int hf){
  return ((v >> 2) << 4) + (hf << 3) + ((v & 3) << 1);
}

__device__ __forceinline__ v16bf load_frag_bf(const bf16* __restrict__ row, int kb, int hf){
  v16bf f;
#pragma unroll
  for (int v = 0; v < 8; ++v){
    int k = kb + kfoff(v, hf);
    f[2*v]   = row[k];
    f[2*v+1] = row[k+1];
  }
  return f;
}

__device__ __forceinline__ v16bf load_frag_f32(const float* __restrict__ row, int kb, int hf){
  v16bf f;
#pragma unroll
  for (int v = 0; v < 8; ++v){
    int k = kb + kfoff(v, hf);
    f[2*v]   = f2bf(row[k]);
    f[2*v+1] = f2bf(row[k+1]);
  }
  return f;
}

__device__ __forceinline__ v8f wmma_bf16(v16bf a, v16bf b, v8f c){
  return __builtin_amdgcn_wmma_f32_16x16x32_bf16(false, a, false, b, (short)0, c, false, false);
}

// ---------------------------------------------------------------------------
// One-time kernels
// ---------------------------------------------------------------------------

__global__ void k_cvt_bf16(const float* __restrict__ src, bf16* __restrict__ dst,
                           int srcRows, int srcK, int dstRows, int dstK){
  int i = blockIdx.x * blockDim.x + threadIdx.x;
  int total = dstRows * dstK;
  if (i >= total) return;
  int r = i / dstK, k = i - r * dstK;
  float v = (r < srcRows && k < srcK) ? src[(size_t)r * srcK + k] : 0.0f;
  dst[i] = f2bf(v);
}

__global__ void k_emb(const float* __restrict__ emb, const int* __restrict__ formulas,
                      bf16* __restrict__ emb_all){
  int i = blockIdx.x * blockDim.x + threadIdx.x;
  int total = NT * NB * NE;
  if (i >= total) return;
  int t = i / (NB * NE); int rem = i - t * NB * NE;
  int b = rem / NE;      int e = rem - b * NE;
  int tok = formulas[b * NT + t];
  emb_all[i] = f2bf(emb[(size_t)tok * NE + e]);
}

__global__ void k_mean(const float* __restrict__ enc, float* __restrict__ meanv){
  int i = blockIdx.x * blockDim.x + threadIdx.x;
  if (i >= NB * NC) return;
  int b = i >> 9, c = i & 511;
  const float* p = enc + (size_t)b * NL * NC + c;
  float acc = 0.0f;
  for (int l = 0; l < NL; ++l) acc += p[(size_t)l * NC];
  meanv[i] = acc * (1.0f / (float)NL);
}

__global__ void k_init(const float* __restrict__ meanv,
                       const float* __restrict__ whw, const float* __restrict__ whb,
                       const float* __restrict__ wcw, const float* __restrict__ wcb,
                       const float* __restrict__ wow, const float* __restrict__ wob,
                       bf16* __restrict__ hbf0, float* __restrict__ cst, bf16* __restrict__ obf){
  int i = blockIdx.x * blockDim.x + threadIdx.x;
  if (i >= 3 * NB * NH) return;
  int which = i / (NB * NH); int rem = i - which * NB * NH;
  int b = rem >> 9, n = rem & 511;
  const float* w; const float* bias;
  if (which == 0)      { w = whw; bias = whb; }
  else if (which == 1) { w = wcw; bias = wcb; }
  else                 { w = wow; bias = wob; }
  const float* m  = meanv + (size_t)b * NC;
  const float* wr = w + (size_t)n * NC;
  float acc = bias[n];
  for (int c = 0; c < NC; ++c) acc += m[c] * wr[c];
  float v = fast_tanh(acc);
  size_t idx = (size_t)b * NH + n;
  if (which == 0)      hbf0[idx] = f2bf(v);
  else if (which == 1) cst[idx]  = v;
  else                 obf[idx]  = f2bf(v);
}

// enc_proj = enc(65536x512) @ W1^T(512x512). Wave: one 16-row M tile x 4 N tiles.
__global__ __launch_bounds__(256) void k_encproj(const float* __restrict__ enc,
                                                 const bf16* __restrict__ w1bf,
                                                 float* __restrict__ encproj){
  int tid  = blockIdx.x * 256 + threadIdx.x;
  int wv   = tid >> 5, lane = tid & 31;
  int mt   = wv >> 3, ng = wv & 7;
  int m0   = mt << 4;
  int hf   = lane >> 4, lm = lane & 15;
  const float* arow = enc + (size_t)(m0 + lm) * NC;
  const bf16* brow[4];
#pragma unroll
  for (int s = 0; s < 4; ++s) brow[s] = w1bf + (size_t)(ng * 64 + s * 16 + lm) * NC;
  v8f acc[4];
#pragma unroll
  for (int s = 0; s < 4; ++s)
#pragma unroll
    for (int r = 0; r < 8; ++r) acc[s][r] = 0.0f;
  for (int kt = 0; kt < 16; ++kt){
    int kb = kt * 32;
    v16bf a = load_frag_f32(arow, kb, hf);
#pragma unroll
    for (int s = 0; s < 4; ++s){
      v16bf bb = load_frag_bf(brow[s], kb, hf);
      acc[s] = wmma_bf16(a, bb, acc[s]);
    }
  }
  int mh = m0 + (hf << 3);
#pragma unroll
  for (int s = 0; s < 4; ++s){
    int n0 = ng * 64 + s * 16;
#pragma unroll
    for (int r = 0; r < 8; ++r)
      encproj[(size_t)(mh + r) * NC + n0 + lm] = acc[s][r];
  }
}

// ---------------------------------------------------------------------------
// Per-step kernels
// ---------------------------------------------------------------------------

__device__ __forceinline__ bf16 x_elem(const bf16* __restrict__ embrow,
                                       const bf16* __restrict__ orow, int kk){
  if (kk < NE)        return embrow[kk];
  if (kk < NE + NH)   return orow[kk - NE];
  return (bf16)0.0f;
}

// Fused gates GEMM + LSTM pointwise, K-split across 5 waves per block.
// Block -> (mt, jt): owns gate tiles n = g*512 + 16*jt for g in {i,f,g,o}.
// 35 unified K tiles (19 x-part + 16 h-part); wave kw owns tiles
// [7*kw, 7*kw+7). Partial C fragments reduced through LDS; wave 0 adds the
// biases and runs the LSTM pointwise (it owns c/h elements -> no races).
__global__ __launch_bounds__(160) void k_lstm(
    const bf16* __restrict__ emb_all, const bf16* __restrict__ obf,
    const bf16* __restrict__ hbf_in,
    const bf16* __restrict__ wih, const bf16* __restrict__ whh,
    const float* __restrict__ bih, const float* __restrict__ bhh,
    float* __restrict__ cst, bf16* __restrict__ hbf_out, int t){
  __shared__ float red[5][4][32][8];   // 20 KB (320 KB/WGP on CDNA5 -> cheap)
  int mt = blockIdx.x >> 5, jt = blockIdx.x & 31;
  int kw = threadIdx.x >> 5, lane = threadIdx.x & 31;
  int m0 = mt << 4, j0 = jt << 4;
  int hf = lane >> 4, lm = lane & 15;
  int ba = m0 + lm;
  const bf16* embrow = emb_all + ((size_t)t * NB + ba) * NE;
  const bf16* orow   = obf + (size_t)ba * NH;
  const bf16* hrow   = hbf_in + (size_t)ba * NH;

  v8f acc[4];
#pragma unroll
  for (int g = 0; g < 4; ++g)
#pragma unroll
    for (int r = 0; r < 8; ++r) acc[g][r] = 0.0f;

  for (int i = 0; i < 7; ++i){
    int kt = kw * 7 + i;             // uniform per wave -> EXEC stays full
    if (kt < 19){                    // x = [emb | o | pad], K padded to 608
      int kb = kt * 32;
      v16bf a;
#pragma unroll
      for (int v = 0; v < 8; ++v){
        int k0 = kb + kfoff(v, hf);
        a[2*v]   = x_elem(embrow, orow, k0);
        a[2*v+1] = x_elem(embrow, orow, k0 + 1);
      }
#pragma unroll
      for (int g = 0; g < 4; ++g){
        const bf16* brow = wih + (size_t)(g * NH + j0 + lm) * 608;
        v16bf bb = load_frag_bf(brow, kb, hf);
        acc[g] = wmma_bf16(a, bb, acc[g]);
      }
    } else {                         // h part, K = 512
      int kb = (kt - 19) * 32;
      v16bf a = load_frag_bf(hrow, kb, hf);
#pragma unroll
      for (int g = 0; g < 4; ++g){
        const bf16* brow = whh + (size_t)(g * NH + j0 + lm) * NH;
        v16bf bb = load_frag_bf(brow, kb, hf);
        acc[g] = wmma_bf16(a, bb, acc[g]);
      }
    }
  }

#pragma unroll
  for (int g = 0; g < 4; ++g)
#pragma unroll
    for (int r = 0; r < 8; ++r) red[kw][g][lane][r] = acc[g][r];
  __syncthreads();

  if (kw == 0){
    float bias[4];
#pragma unroll
    for (int g = 0; g < 4; ++g){
      int n = g * NH + j0 + lm;
      bias[g] = bih[n] + bhh[n];
    }
    int j = j0 + lm;
#pragma unroll
    for (int r = 0; r < 8; ++r){
      float gs[4];
#pragma unroll
      for (int g = 0; g < 4; ++g){
        float s = bias[g];
#pragma unroll
        for (int w = 0; w < 5; ++w) s += red[w][g][lane][r];
        gs[g] = s;
      }
      int b = m0 + (hf << 3) + r;
      size_t idx = (size_t)b * NH + j;
      float cold = cst[idx];
      float cn = fast_sigmoid(gs[1]) * cold + fast_sigmoid(gs[0]) * fast_tanh(gs[2]);
      float hn = fast_sigmoid(gs[3]) * fast_tanh(cn);
      cst[idx] = cn;
      hbf_out[idx] = f2bf(hn);
    }
  }
}

// hW2 = h @ W2^T (64x512, K=512). Block -> (mt, ng); 4 waves split 16 K tiles.
__global__ __launch_bounds__(128) void k_hw2(const bf16* __restrict__ hbf,
                                             const bf16* __restrict__ w2bf,
                                             float* __restrict__ hw2){
  __shared__ float red[4][4][32][8];   // 16 KB
  int mt = blockIdx.x >> 3, ng = blockIdx.x & 7;
  int kw = threadIdx.x >> 5, lane = threadIdx.x & 31;
  int m0 = mt << 4;
  int hf = lane >> 4, lm = lane & 15;
  const bf16* arow = hbf + (size_t)(m0 + lm) * NH;
  v8f acc[4];
#pragma unroll
  for (int s = 0; s < 4; ++s)
#pragma unroll
    for (int r = 0; r < 8; ++r) acc[s][r] = 0.0f;
  for (int i = 0; i < 4; ++i){
    int kt = kw * 4 + i;
    int kb = kt * 32;
    v16bf a = load_frag_bf(arow, kb, hf);
#pragma unroll
    for (int s = 0; s < 4; ++s){
      const bf16* brow = w2bf + (size_t)(ng * 64 + s * 16 + lm) * NH;
      v16bf bb = load_frag_bf(brow, kb, hf);
      acc[s] = wmma_bf16(a, bb, acc[s]);
    }
  }
#pragma unroll
  for (int s = 0; s < 4; ++s)
#pragma unroll
    for (int r = 0; r < 8; ++r) red[kw][s][lane][r] = acc[s][r];
  __syncthreads();
  if (kw == 0){
    int mh = m0 + (hf << 3);
#pragma unroll
    for (int s = 0; s < 4; ++s){
      int n0 = ng * 64 + s * 16;
#pragma unroll
      for (int r = 0; r < 8; ++r){
        float v = 0.0f;
#pragma unroll
        for (int w = 0; w < 4; ++w) v += red[w][s][lane][r];
        hw2[(size_t)(mh + r) * NC + n0 + lm] = v;
      }
    }
  }
}

// scores[b][l] = sum_c tanh(enc_proj[b][l][c] + hW2[b][c]) * beta[c].
// One wave per (b,l); the 134 MB/step HBM stream; float4 coalesced loads.
__global__ void k_scores(const float* __restrict__ encproj, const float* __restrict__ hw2,
                         const float* __restrict__ beta, float* __restrict__ scores){
  int tid = blockIdx.x * 256 + threadIdx.x;
  int wv  = tid >> 5, lane = tid & 31;
  int b   = wv >> 10, l = wv & 1023;
  const float4* ep = (const float4*)(encproj + ((size_t)b * NL + l) * NC);
  const float4* hw = (const float4*)(hw2 + (size_t)b * NC);
  const float4* be = (const float4*)beta;
  float acc = 0.0f;
#pragma unroll
  for (int q = 0; q < 4; ++q){
    int idx = lane + 32 * q;
    float4 e  = ep[idx];
    float4 h  = hw[idx];
    float4 bt = be[idx];
    acc += fast_tanh(e.x + h.x) * bt.x + fast_tanh(e.y + h.y) * bt.y
         + fast_tanh(e.z + h.z) * bt.z + fast_tanh(e.w + h.w) * bt.w;
  }
#pragma unroll
  for (int off = 16; off; off >>= 1) acc += __shfl_xor(acc, off, 32);
  if (lane == 0) scores[(size_t)b * NL + l] = acc;
}

__global__ void k_softmax(const float* __restrict__ scores, float* __restrict__ alpha){
  __shared__ float red[256];
  int b = blockIdx.x, tid = threadIdx.x;
  const float* s = scores + (size_t)b * NL;
  float m = -3.0e38f;
#pragma unroll
  for (int q = 0; q < 4; ++q) m = fmaxf(m, s[tid + 256 * q]);
  red[tid] = m; __syncthreads();
  for (int o = 128; o; o >>= 1){ if (tid < o) red[tid] = fmaxf(red[tid], red[tid + o]); __syncthreads(); }
  m = red[0]; __syncthreads();
  float e[4]; float sum = 0.0f;
#pragma unroll
  for (int q = 0; q < 4; ++q){ e[q] = fast_exp(s[tid + 256 * q] - m); sum += e[q]; }
  red[tid] = sum; __syncthreads();
  for (int o = 128; o; o >>= 1){ if (tid < o) red[tid] += red[tid + o]; __syncthreads(); }
  float inv = fast_rcp(red[0]);
#pragma unroll
  for (int q = 0; q < 4; ++q) alpha[(size_t)b * NL + tid + 256 * q] = e[q] * inv;
}

// context partials: block (b, lc) covers 64 L positions; thread owns 2 columns.
__global__ void k_ctxpart(const float* __restrict__ enc, const float* __restrict__ alpha,
                          float* __restrict__ part){
  int b = blockIdx.x, lc = blockIdx.y, tid = threadIdx.x;
  int c0 = tid, c1 = tid + 256;
  const float* al = alpha + (size_t)b * NL + lc * 64;
  const float* e  = enc + ((size_t)b * NL + lc * 64) * NC;
  float a0 = 0.0f, a1 = 0.0f;
  for (int l = 0; l < 64; ++l){
    float w = al[l];
    a0 += w * e[(size_t)l * NC + c0];
    a1 += w * e[(size_t)l * NC + c1];
  }
  float* p = part + ((size_t)lc * NB + b) * NC;
  p[c0] = a0; p[c1] = a1;
}

// deterministic reduce of 16 partials -> bf16 context
__global__ void k_ctxred(const float* __restrict__ part, bf16* __restrict__ ctxbf){
  int i = blockIdx.x * blockDim.x + threadIdx.x;
  if (i >= NB * NC) return;
  int b = i >> 9, c = i & 511;
  float acc = 0.0f;
  for (int lc = 0; lc < 16; ++lc) acc += part[((size_t)lc * NB + b) * NC + c];
  ctxbf[i] = f2bf(acc);
}

// o = tanh([h | ctx] @ W3^T), K = 1024. Block -> (mt, ng); 4 waves x 8 K tiles.
__global__ __launch_bounds__(128) void k_o(const bf16* __restrict__ hbf,
                                           const bf16* __restrict__ ctxbf,
                                           const bf16* __restrict__ w3bf,
                                           bf16* __restrict__ obf){
  __shared__ float red[4][4][32][8];   // 16 KB
  int mt = blockIdx.x >> 3, ng = blockIdx.x & 7;
  int kw = threadIdx.x >> 5, lane = threadIdx.x & 31;
  int m0 = mt << 4;
  int hf = lane >> 4, lm = lane & 15;
  int ba = m0 + lm;
  const bf16* hrow = hbf + (size_t)ba * NH;
  const bf16* crow = ctxbf + (size_t)ba * NC;
  v8f acc[4];
#pragma unroll
  for (int s = 0; s < 4; ++s)
#pragma unroll
    for (int r = 0; r < 8; ++r) acc[s][r] = 0.0f;
  for (int i = 0; i < 8; ++i){
    int kt = kw * 8 + i;
    int kb = kt * 32;
    v16bf a;
#pragma unroll
    for (int v = 0; v < 8; ++v){
      int k0 = kb + kfoff(v, hf);
      a[2*v]   = (k0     < NH) ? hrow[k0]     : crow[k0 - NH];
      a[2*v+1] = (k0 + 1 < NH) ? hrow[k0 + 1] : crow[k0 + 1 - NH];
    }
#pragma unroll
    for (int s = 0; s < 4; ++s){
      const bf16* brow = w3bf + (size_t)(ng * 64 + s * 16 + lm) * (NH + NC);
      v16bf bb = load_frag_bf(brow, kb, hf);
      acc[s] = wmma_bf16(a, bb, acc[s]);
    }
  }
#pragma unroll
  for (int s = 0; s < 4; ++s)
#pragma unroll
    for (int r = 0; r < 8; ++r) red[kw][s][lane][r] = acc[s][r];
  __syncthreads();
  if (kw == 0){
    int mh = m0 + (hf << 3);
#pragma unroll
    for (int s = 0; s < 4; ++s){
      int n0 = ng * 64 + s * 16;
#pragma unroll
      for (int r = 0; r < 8; ++r){
        float v = 0.0f;
#pragma unroll
        for (int w = 0; w < 4; ++w) v += red[w][s][lane][r];
        obf[(size_t)(mh + r) * NH + n0 + lm] = f2bf(fast_tanh(v));
      }
    }
  }
}

// Fused logits GEMM (64x1008, K=512) + row softmax over V=1000. Block per
// 16-batch tile; 16x1008 fp32 tile lives in LDS (63 KB).
__global__ __launch_bounds__(1024) void k_logits(const bf16* __restrict__ obf,
                                                 const bf16* __restrict__ woutbf,
                                                 float* __restrict__ outp, int t){
  __shared__ float lg[16 * 1008];
  int mt = blockIdx.x, m0 = mt << 4;
  int tid = threadIdx.x, lane = tid & 31, wv = tid >> 5;
  int hf = lane >> 4, lm = lane & 15;
  const bf16* arow = obf + (size_t)(m0 + lm) * NH;
  for (int s = 0; s < 2; ++s){
    int nt = wv * 2 + s;
    if (nt >= 63) break;
    int n0 = nt * 16;
    const bf16* brow = woutbf + (size_t)(n0 + lm) * NH;
    v8f acc;
#pragma unroll
    for (int r = 0; r < 8; ++r) acc[r] = 0.0f;
    for (int kt = 0; kt < 16; ++kt){
      int kb = kt * 32;
      v16bf a  = load_frag_bf(arow, kb, hf);
      v16bf bb = load_frag_bf(brow, kb, hf);
      acc = wmma_bf16(a, bb, acc);
    }
    int mh = hf << 3;
#pragma unroll
    for (int r = 0; r < 8; ++r) lg[(mh + r) * 1008 + n0 + lm] = acc[r];
  }
  __syncthreads();
  if (wv < 16){
    int row = wv;
    float* p = lg + row * 1008;
    float mx = -3.0e38f;
    for (int i = lane; i < NV; i += 32) mx = fmaxf(mx, p[i]);
#pragma unroll
    for (int off = 16; off; off >>= 1) mx = fmaxf(mx, __shfl_xor(mx, off, 32));
    float sum = 0.0f;
    for (int i = lane; i < NV; i += 32){ float e = fast_exp(p[i] - mx); p[i] = e; sum += e; }
#pragma unroll
    for (int off = 16; off; off >>= 1) sum += __shfl_xor(sum, off, 32);
    float inv = fast_rcp(sum);
    int b = m0 + row;
    float* dst = outp + ((size_t)b * NT + t) * NV;
    for (int i = lane; i < NV; i += 32) dst[i] = p[i] * inv;
  }
}

// ---------------------------------------------------------------------------
// Host launcher
// ---------------------------------------------------------------------------
extern "C" void kernel_launch(void* const* d_in, const int* in_sizes, int n_in,
                              void* d_out, int out_size, void* d_ws, size_t ws_size,
                              hipStream_t stream){
  const float* enc      = (const float*)d_in[0];
  const int*   formulas = (const int*)  d_in[1];
  const float* emb      = (const float*)d_in[2];
  const float* W_ih     = (const float*)d_in[3];
  const float* W_hh     = (const float*)d_in[4];
  const float* b_ih     = (const float*)d_in[5];
  const float* b_hh     = (const float*)d_in[6];
  const float* wh_w     = (const float*)d_in[7];
  const float* wh_b     = (const float*)d_in[8];
  const float* wc_w     = (const float*)d_in[9];
  const float* wc_b     = (const float*)d_in[10];
  const float* wo_w     = (const float*)d_in[11];
  const float* wo_b     = (const float*)d_in[12];
  const float* beta     = (const float*)d_in[13];
  const float* W1       = (const float*)d_in[14];
  const float* W2       = (const float*)d_in[15];
  const float* W3       = (const float*)d_in[16];
  const float* W_out    = (const float*)d_in[17];
  float* out = (float*)d_out;

  char* ws = (char*)d_ws;
  size_t off = 0;
  auto carve = [&](size_t bytes) -> char* {
    char* p = ws + off;
    off += (bytes + 255) & ~(size_t)255;
    return p;
  };
  float* encproj = (float*)carve((size_t)NB * NL * NC * 4);      // 134 MB
  bf16*  wihbf   = (bf16*) carve((size_t)2048 * 608 * 2);
  bf16*  whhbf   = (bf16*) carve((size_t)2048 * 512 * 2);
  bf16*  w2bf    = (bf16*) carve((size_t)512 * 512 * 2);
  bf16*  w3bf    = (bf16*) carve((size_t)512 * 1024 * 2);
  bf16*  woutbf  = (bf16*) carve((size_t)1008 * 512 * 2);
  bf16*  w1bf    = (bf16*) carve((size_t)512 * 512 * 2);
  bf16*  emball  = (bf16*) carve((size_t)NT * NB * NE * 2);
  bf16*  hbuf    = (bf16*) carve((size_t)2 * NB * NH * 2);       // double buffer
  bf16*  obf     = (bf16*) carve((size_t)NB * NH * 2);
  bf16*  ctxbf   = (bf16*) carve((size_t)NB * NC * 2);
  float* cst     = (float*)carve((size_t)NB * NH * 4);
  float* hw2     = (float*)carve((size_t)NB * NC * 4);
  float* scores  = (float*)carve((size_t)NB * NL * 4);
  float* alpha   = (float*)carve((size_t)NB * NL * 4);
  float* ctxpart = (float*)carve((size_t)16 * NB * NC * 4);
  float* meanv   = (float*)carve((size_t)NB * NC * 4);
  (void)in_sizes; (void)n_in; (void)out_size; (void)ws_size;

  // one-time: weight conversion + padding
  k_cvt_bf16<<<(2048 * 608 + 255) / 256, 256, 0, stream>>>(W_ih,  wihbf,  2048, 592, 2048, 608);
  k_cvt_bf16<<<(2048 * 512 + 255) / 256, 256, 0, stream>>>(W_hh,  whhbf,  2048, 512, 2048, 512);
  k_cvt_bf16<<<(512 * 512 + 255)  / 256, 256, 0, stream>>>(W2,    w2bf,   512,  512, 512,  512);
  k_cvt_bf16<<<(512 * 1024 + 255) / 256, 256, 0, stream>>>(W3,    w3bf,   512, 1024, 512, 1024);
  k_cvt_bf16<<<(1008 * 512 + 255) / 256, 256, 0, stream>>>(W_out, woutbf, 1000, 512, 1008, 512);
  k_cvt_bf16<<<(512 * 512 + 255)  / 256, 256, 0, stream>>>(W1,    w1bf,   512,  512, 512,  512);
  // one-time: embeddings, mean, initial state, enc_proj
  k_emb<<<(NT * NB * NE + 255) / 256, 256, 0, stream>>>(emb, formulas, emball);
  k_mean<<<(NB * NC + 255) / 256, 256, 0, stream>>>(enc, meanv);
  k_init<<<(3 * NB * NH + 255) / 256, 256, 0, stream>>>(meanv, wh_w, wh_b, wc_w, wc_b,
                                                        wo_w, wo_b, hbuf, cst, obf);
  k_encproj<<<4096, 256, 0, stream>>>(enc, w1bf, encproj);

  for (int t = 0; t < NT; ++t){
    const bf16* hin = hbuf + (size_t)(t & 1) * NB * NH;
    bf16* hout      = hbuf + (size_t)((t + 1) & 1) * NB * NH;
    k_lstm<<<128, 160, 0, stream>>>(emball, obf, hin, wihbf, whhbf, b_ih, b_hh, cst, hout, t);
    k_hw2<<<32, 128, 0, stream>>>(hout, w2bf, hw2);
    k_scores<<<8192, 256, 0, stream>>>(encproj, hw2, beta, scores);
    k_softmax<<<NB, 256, 0, stream>>>(scores, alpha);
    k_ctxpart<<<dim3(NB, 16), 256, 0, stream>>>(enc, alpha, ctxpart);
    k_ctxred<<<(NB * NC + 255) / 256, 256, 0, stream>>>(ctxpart, ctxbf);
    k_o<<<32, 128, 0, stream>>>(hout, ctxbf, w3bf, obf);
    k_logits<<<4, 1024, 0, stream>>>(obf, woutbf, out, t);
  }
}
